// RNN_2095944040499
// MI455X (gfx1250) — compile-verified, matrix operand order checked by reading, not run on hardware
//
#include <hip/hip_runtime.h>
#include <hip/hip_bf16.h>

// ---- problem constants ----
#define BB 64
#define SS 2048
#define HH 512
#define II 48
#define OO 49
#define ALPHA 0.2f
#define NOISE_STD 0.005f
#define TAU_MAX 5.0f
#define TAU_MIN 0.2f

typedef __bf16 bf16;
typedef __attribute__((ext_vector_type(16))) __bf16 v16bf;
typedef __attribute__((ext_vector_type(8)))  __bf16 bf16x8;
typedef __attribute__((ext_vector_type(8)))  float  v8f;
typedef __attribute__((ext_vector_type(4)))  float  f32x4;

// workspace layout (bytes)
#define WS_JBF    0          // 512*512 bf16   = 524288
#define WS_WIBT   524288     // 512*64  bf16   =  65536  (N-major, K padded 48->64)
#define WS_WOUTT  589824     // 64*512  bf16   =  65536  (N-major, O padded 49->64)
#define WS_VEC    655360     // 4*512 f32: inv_tau | eg | b | h0

__device__ __forceinline__ float softplus_f(float x) {
    return fmaxf(x, 0.f) + __logf(1.f + __expf(-fabsf(x)));
}

// Load one lane's 16 bf16 A/B fragment: two contiguous 16B chunks at
// row_base + off and row_base + off + 16 (elements).
__device__ __forceinline__ v16bf ld_frag(const bf16* row_base, int off) {
    bf16x8 lo = *(const bf16x8*)(row_base + off);
    bf16x8 hi = *(const bf16x8*)(row_base + off + 16);
    return __builtin_shufflevector(lo, hi, 0,1,2,3,4,5,6,7,8,9,10,11,12,13,14,15);
}

__device__ __forceinline__ v8f wmma_bf16(v16bf a, v16bf b, v8f c) {
    return __builtin_amdgcn_wmma_f32_16x16x32_bf16(false, a, false, b, (short)0, c,
                                                   false, false);
}

// ---------------- prep: build bf16 weights + per-unit vectors ----------------
__global__ void rnn_prep(const float* __restrict__ wrec, const float* __restrict__ mwrec,
                         const float* __restrict__ wi,   const float* __restrict__ si,
                         const float* __restrict__ wout, const float* __restrict__ b,
                         const float* __restrict__ g,    const float* __restrict__ taus,
                         const float* __restrict__ h0,
                         bf16* __restrict__ Jbf, bf16* __restrict__ wibT,
                         bf16* __restrict__ woutT, float* __restrict__ vec) {
    int i = blockIdx.x * blockDim.x + threadIdx.x;
    // Jmat row-major bf16: Jbf[n*512+k] = exp(wrec[n,k])*mwrec[n,k]
    // (== N-major B layout for z = a @ Jmat^T)
    if (i < HH * HH) Jbf[i] = (bf16)(__expf(wrec[i]) * mwrec[i]);
    // wibT[h][k] = exp(si[k]) * wi[k,h], K padded to 64 with zeros
    if (i < HH * 64) {
        int h = i >> 6, k = i & 63;
        wibT[i] = (k < II) ? (bf16)(__expf(si[k]) * wi[k * HH + h]) : (bf16)0.f;
    }
    // woutT[o][k] = wout[k,o], O padded to 64 with zeros
    if (i < 64 * HH) {
        int o = i >> 9, k = i & 511;
        woutT[i] = (o < OO) ? (bf16)(wout[k * OO + o]) : (bf16)0.f;
    }
    if (i < HH) {
        vec[i]            = 1.f / (0.5f * (TAU_MAX + TAU_MIN)
                                 + 0.5f * (TAU_MAX - TAU_MIN) * tanhf(taus[i]));
        vec[HH + i]       = __expf(g[i]);
        vec[2 * HH + i]   = b[i];
        vec[3 * HH + i]   = h0[i];
    }
}

// ---------------- main: persistent per-batch-chunk recurrence ----------------
// LDS budget (320KB/WGP on CDNA5): aS 16KB + xS 2KB + nzS 32KB + jS 256KB = 306KB
// -> one workgroup per WGP, which is exactly the occupancy we want (grid = 4).
__global__ __launch_bounds__(512, 1)
void rnn_main(const float* __restrict__ x, const float* __restrict__ noise,
              const bf16* __restrict__ Jbf, const bf16* __restrict__ wibT,
              const bf16* __restrict__ woutT, const float* __restrict__ vec,
              float* __restrict__ out) {
    __shared__ __align__(16) bf16  aS[16 * HH];    // eg*r staged as bf16 (A matrix)
    __shared__ __align__(16) bf16  xS[16 * 64];    // x_t staged bf16, K padded to 64
    __shared__ __align__(16) float nzS[16 * HH];   // noise_t staged fp32
    __shared__ __align__(16) bf16  jS[256 * HH];   // even N-tiles of Jmat, resident

    const int tid   = threadIdx.x;
    const int lane  = tid & 31;
    const int wid   = tid >> 5;          // 16 waves
    const int lh    = lane & 15;         // A-row (batch) AND B-row (N) index in tile
    const int hihalf= lane >> 4;
    const int mbase = 8 * hihalf;        // C/D m offset
    const int b0    = blockIdx.x * 16;   // batch rows [b0, b0+16)

    // ---- one-time: cache even N-tiles of Jmat in LDS ----
    // global row n (tile = n>>4 even) -> LDS slot ((n>>5)<<4)|(n&15)
    for (int c = tid; c < 256 * (HH / 8); c += 512) {    // 8-elem (16B) chunks
        int slot = c >> 6;                               // 64 chunks per row
        int col  = (c & 63) * 8;
        int n    = ((slot >> 4) << 5) | (slot & 15);
        *(bf16x8*)(jS + slot * HH + col) = *(const bf16x8*)(Jbf + (long)n * HH + col);
    }

    // each wave owns N-tiles {2*wid, 2*wid+1} -> columns [32*wid, 32*wid+32)
    const int n0 = 32 * wid + lh;        // even tile  -> B rows from LDS
    const int n1 = n0 + 16;              // odd tile   -> B rows from L2
    const float invt0 = vec[n0],          invt1 = vec[n1];
    const float eg0   = vec[HH + n0],     eg1   = vec[HH + n1];
    const float bb0   = vec[2 * HH + n0], bb1   = vec[2 * HH + n1];
    const float h00   = vec[3 * HH + n0], h01   = vec[3 * HH + n1];

    v8f h_0, h_1;
    #pragma unroll
    for (int v = 0; v < 8; ++v) { h_0[v] = h00; h_1[v] = h01; }

    const int o = 16 * wid + lh;             // readout column (waves 0..3 only)
    const long SO = (long)SS * OO;

    __syncthreads();                         // jS resident

    for (int t = 0; t < SS; ++t) {
        // ---- activation: r = softplus(h + b), a = eg * r ----
        v8f a0, a1;
        #pragma unroll
        for (int v = 0; v < 8; ++v) {
            a0[v] = eg0 * softplus_f(h_0[v] + bb0);
            a1[v] = eg1 * softplus_f(h_1[v] + bb1);
        }
        __syncthreads();                     // previous step's LDS reads done
        #pragma unroll
        for (int v = 0; v < 8; ++v) {        // scatter a into A-staging (own cols)
            aS[(mbase + v) * HH + n0] = (bf16)a0[v];
            aS[(mbase + v) * HH + n1] = (bf16)a1[v];
        }
        if (t < SS - 1) {
            // stage noise[:, t] : 16x512 f32, coalesced float4
            #pragma unroll
            for (int q = 0; q < 4; ++q) {
                int fi = (q * 512 + tid) * 4;
                int m = fi >> 9, hh = fi & 511;
                f32x4 nv = *(const f32x4*)(noise + (((long)(b0 + m) * SS + t) * HH + hh));
                *(f32x4*)(nzS + fi) = nv;
            }
            // stage x[:, t] : 16x48 -> 16x64 bf16 (zero pad)
            for (int i2 = tid; i2 < 16 * 64; i2 += 512) {
                int m = i2 >> 6, c = i2 & 63;
                xS[i2] = (c < II) ? (bf16)x[(((long)(b0 + m) * SS + t) * II + c)]
                                  : (bf16)0.f;
            }
        }
        __syncthreads();

        // ---- readout out_t = a @ wout  (waves 0..3, one 16-wide O tile each) ----
        if (wid < 4) {
            v8f oacc = {};
            const bf16* brow = woutT + (long)o * HH;
            const bf16* arow = aS + lh * HH;
            #pragma unroll 4
            for (int kt = 0; kt < 16; ++kt) {
                int kb = kt * 32 + 8 * hihalf;
                oacc = wmma_bf16(ld_frag(arow, kb), ld_frag(brow, kb), oacc);
            }
            if (o < OO) {
                float* op = out + (long)(b0 + mbase) * SO + (long)t * OO + o;
                #pragma unroll
                for (int v = 0; v < 8; ++v) op[(long)v * SO] = oacc[v];
            }
        }

        if (t == SS - 1) break;

        // ---- z = x_t @ (exp(si)*wi)  +  a @ Jmat^T ----
        v8f z0 = {}, z1 = {};
        {
            const bf16* w0 = wibT + (long)n0 * 64;
            const bf16* w1 = wibT + (long)n1 * 64;
            const bf16* xrow = xS + lh * 64;
            #pragma unroll
            for (int kt = 0; kt < 2; ++kt) {
                int kb = kt * 32 + 8 * hihalf;
                v16bf av = ld_frag(xrow, kb);
                z0 = wmma_bf16(av, ld_frag(w0, kb), z0);
                z1 = wmma_bf16(av, ld_frag(w1, kb), z1);
            }
            // even tile's B rows come from LDS-resident jS, odd tile's from L2
            const bf16* j0 = jS + ((wid << 4) | lh) * HH;
            const bf16* j1 = Jbf + (long)n1 * HH;
            const bf16* arow = aS + lh * HH;
            #pragma unroll 4
            for (int kt = 0; kt < 16; ++kt) {
                int kb = kt * 32 + 8 * hihalf;
                v16bf av = ld_frag(arow, kb);
                z0 = wmma_bf16(av, ld_frag(j0, kb), z0);
                z1 = wmma_bf16(av, ld_frag(j1, kb), z1);
            }
        }

        // ---- h update ----
        #pragma unroll
        for (int v = 0; v < 8; ++v) {
            float nz0 = nzS[(mbase + v) * HH + n0];
            float nz1 = nzS[(mbase + v) * HH + n1];
            h_0[v] += NOISE_STD * nz0 + ALPHA * invt0 * (z0[v] - h_0[v]);
            h_1[v] += NOISE_STD * nz1 + ALPHA * invt1 * (z1[v] - h_1[v]);
        }
    }
}

extern "C" void kernel_launch(void* const* d_in, const int* in_sizes, int n_in,
                              void* d_out, int out_size, void* d_ws, size_t ws_size,
                              hipStream_t stream) {
    (void)in_sizes; (void)n_in; (void)out_size; (void)ws_size;
    const float* x     = (const float*)d_in[0];
    const float* noise = (const float*)d_in[1];
    const float* wi    = (const float*)d_in[2];
    const float* si    = (const float*)d_in[3];
    const float* wrec  = (const float*)d_in[4];
    const float* mwrec = (const float*)d_in[5];
    const float* wout  = (const float*)d_in[6];
    const float* b     = (const float*)d_in[7];
    const float* g     = (const float*)d_in[8];
    const float* taus  = (const float*)d_in[9];
    const float* h0    = (const float*)d_in[10];

    char* ws = (char*)d_ws;
    bf16*  Jbf   = (bf16*)(ws + WS_JBF);
    bf16*  wibT  = (bf16*)(ws + WS_WIBT);
    bf16*  woutT = (bf16*)(ws + WS_WOUTT);
    float* vec   = (float*)(ws + WS_VEC);

    rnn_prep<<<(HH * HH + 255) / 256, 256, 0, stream>>>(
        wrec, mwrec, wi, si, wout, b, g, taus, h0, Jbf, wibT, woutT, vec);

    rnn_main<<<BB / 16, 512, 0, stream>>>(
        x, noise, Jbf, wibT, woutT, vec, (float*)d_out);
}
